// Attention_15607911154216
// MI455X (gfx1250) — matrix operand order repaired and checked by reading, not compiled, not run on hardware
//
#include <hip/hip_runtime.h>
#include <hip/hip_bf16.h>

// ---------------------------------------------------------------------------
// ConvS2S attention block on MI455X (gfx1250), wave32 + WMMA bf16.
//   G1: cc[b,t,e]   = (conved^T @ w_h2e^T + b_h2e + x) * scale      (bf16 ws)
//   G2: energy[b,t,s]= cc @ enc_conved^T                            (f32 d_out)
//   SM: softmax over s, in place in d_out
//   G3: att[b,t,e]  = attn @ enc_combined                           (bf16 ws)
//   G4: out2[b,c,t] = conved + (att @ w_e2h^T + b_e2h)^T            (f32 d_out)
// ---------------------------------------------------------------------------

typedef __attribute__((ext_vector_type(16))) __bf16 v16bf;
typedef __attribute__((ext_vector_type(8)))  float  v8f;

#define DEV __device__ __forceinline__

constexpr int NB = 8;     // batch
constexpr int NC = 2048;  // channels
constexpr int NT = 1024;  // decoder time
constexpr int NS = 1024;  // encoder time
constexpr int NE = 1024;  // embed

constexpr int BM = 128, BN = 128, BK = 32;
constexpr int LDT = BK + 16;   // 48 halfwords = 96B row stride (16B multiple)

struct Params {
  const float* conved;        // [B,C,T]
  const float* enc_conved;    // [B,S,E]
  const float* enc_combined;  // [B,S,E]
  const float* x;             // [B,T,E]
  const float* scale;         // [1]
  const float* w_h2e;         // [E,C]
  const float* b_h2e;         // [E]
  const float* w_e2h;         // [C,E]
  const float* b_e2h;         // [C]
  float* attn;                // d_out part 1: [B,T,S]
  float* out2;                // d_out part 2: [B,C,T]
  unsigned short* cc;         // ws: [B,T,E] bf16
  unsigned short* att;        // ws: [B,T,E] bf16
};

DEV unsigned short f2bf(float f) {          // round-to-nearest-even f32 -> bf16
  unsigned int u = __float_as_uint(f);
  u += 0x7FFFu + ((u >> 16) & 1u);
  return (unsigned short)(u >> 16);
}

union Frag { v16bf v; uint4 q[2]; };

// MODE: 0=G1, 1=G2, 2=G3, 3=G4
template <int MODE>
__global__ __launch_bounds__(256) void gemm_wmma(Params p) {
  __shared__ __align__(16) unsigned short lA[BM * LDT];   // [m][k]
  __shared__ __align__(16) unsigned short lB[BN * LDT];   // [n][k]

  const int bz  = blockIdx.z;
  const int m0  = blockIdx.x * BM;
  const int n0  = blockIdx.y * BN;
  const int tid = threadIdx.x;
  const int lane = tid & 31;
  const int wave = tid >> 5;
  const int lh  = lane >> 4;        // lane half (K/M select per ISA layout)
  const int l16 = lane & 15;
  const int wm  = (wave >> 2) * 64; // 2 waves along M
  const int wn  = (wave & 3) * 32;  // 4 waves along N

  const int K = (MODE == 0) ? NC : ((MODE == 2) ? NS : NE);

  const v8f vzero = {0.f, 0.f, 0.f, 0.f, 0.f, 0.f, 0.f, 0.f};
  v8f acc[4][2];
  #pragma unroll
  for (int mi = 0; mi < 4; ++mi)
    #pragma unroll
    for (int ni = 0; ni < 2; ++ni) acc[mi][ni] = vzero;

  const int base = tid * 16;   // 256 thr * 16 elem = 4096 = one 128x32 tile

  for (int k0 = 0; k0 < K; k0 += BK) {
    // ------------------ stage A tile into LDS (bf16) ------------------
    if (MODE == 0) {
      // A[t,c] = conved[b][c*T + t] : contiguous in t, scatter over LDS rows
      const int c_l = base >> 7, t_l = base & 127;
      const float4* s4 =
          (const float4*)(p.conved + ((size_t)bz * NC + k0 + c_l) * NT + m0 + t_l);
      #pragma unroll
      for (int jj = 0; jj < 4; ++jj) {
        float4 v = s4[jj];
        unsigned short* d = &lA[(t_l + jj * 4) * LDT + c_l];
        d[0 * LDT] = f2bf(v.x); d[1 * LDT] = f2bf(v.y);
        d[2 * LDT] = f2bf(v.z); d[3 * LDT] = f2bf(v.w);
      }
    } else if (MODE == 1 || MODE == 3) {
      // A already bf16 row-major in workspace
      const int t_l = base >> 5, e_l = base & 31;
      const unsigned short* src =
          (MODE == 1 ? p.cc : p.att) + ((size_t)bz * NT + m0 + t_l) * NE + k0 + e_l;
      if (k0 + BK < K) __builtin_prefetch(src + BK, 0, 1);   // global_prefetch_b8
      const uint4* s4 = (const uint4*)src;
      uint4* d = (uint4*)&lA[t_l * LDT + e_l];
      d[0] = s4[0]; d[1] = s4[1];
    } else {
      // A[t,s] = attention (f32, in d_out), row-major
      const int t_l = base >> 5, s_l = base & 31;
      const float4* s4 =
          (const float4*)(p.attn + ((size_t)bz * NT + m0 + t_l) * NS + k0 + s_l);
      unsigned short* d = &lA[t_l * LDT + s_l];
      #pragma unroll
      for (int jj = 0; jj < 4; ++jj) {
        float4 v = s4[jj];
        d[jj * 4 + 0] = f2bf(v.x); d[jj * 4 + 1] = f2bf(v.y);
        d[jj * 4 + 2] = f2bf(v.z); d[jj * 4 + 3] = f2bf(v.w);
      }
    }

    // ------------- stage B tile into LDS as [n][k] (bf16) -------------
    if (MODE == 0) {
      // B[c,e] = w_h2e[e*C + c] : rows of 32 c per e
      const int e_l = base >> 5, c_l = base & 31;
      const float4* s4 = (const float4*)(p.w_h2e + (size_t)(n0 + e_l) * NC + k0 + c_l);
      unsigned short* d = &lB[e_l * LDT + c_l];
      #pragma unroll
      for (int jj = 0; jj < 4; ++jj) {
        float4 v = s4[jj];
        d[jj * 4 + 0] = f2bf(v.x); d[jj * 4 + 1] = f2bf(v.y);
        d[jj * 4 + 2] = f2bf(v.z); d[jj * 4 + 3] = f2bf(v.w);
      }
    } else if (MODE == 1) {
      // B[e,s] = enc_conved[s*E + e] : n = s, k = e (contig in e)
      const int s_l = base >> 5, e_l = base & 31;
      const float4* s4 =
          (const float4*)(p.enc_conved + ((size_t)bz * NS + n0 + s_l) * NE + k0 + e_l);
      unsigned short* d = &lB[s_l * LDT + e_l];
      #pragma unroll
      for (int jj = 0; jj < 4; ++jj) {
        float4 v = s4[jj];
        d[jj * 4 + 0] = f2bf(v.x); d[jj * 4 + 1] = f2bf(v.y);
        d[jj * 4 + 2] = f2bf(v.z); d[jj * 4 + 3] = f2bf(v.w);
      }
    } else if (MODE == 2) {
      // B[s,e] = enc_combined row-major : n = e, k = s -> transpose into LDS
      const int s_l = base >> 7, e_l = base & 127;
      const float4* s4 =
          (const float4*)(p.enc_combined + ((size_t)bz * NS + k0 + s_l) * NE + n0 + e_l);
      #pragma unroll
      for (int jj = 0; jj < 4; ++jj) {
        float4 v = s4[jj];
        unsigned short* d = &lB[(e_l + jj * 4) * LDT + s_l];
        d[0 * LDT] = f2bf(v.x); d[1 * LDT] = f2bf(v.y);
        d[2 * LDT] = f2bf(v.z); d[3 * LDT] = f2bf(v.w);
      }
    } else {
      // B[e,c] = w_e2h[c*E + e] : n = c, k = e (contig in e)
      const int c_l = base >> 5, e_l = base & 31;
      const float4* s4 = (const float4*)(p.w_e2h + (size_t)(n0 + c_l) * NE + k0 + e_l);
      unsigned short* d = &lB[c_l * LDT + e_l];
      #pragma unroll
      for (int jj = 0; jj < 4; ++jj) {
        float4 v = s4[jj];
        d[jj * 4 + 0] = f2bf(v.x); d[jj * 4 + 1] = f2bf(v.y);
        d[jj * 4 + 2] = f2bf(v.z); d[jj * 4 + 3] = f2bf(v.w);
      }
    }

    __syncthreads();

    // ---- fragments per ISA VGPR layouts (CDNA5 ISA) ----
    Frag fa[4], fb[2];
    #pragma unroll
    for (int mi = 0; mi < 4; ++mi) {
      const unsigned short* pa = &lA[(wm + mi * 16 + l16) * LDT + lh * 8];
      fa[mi].q[0] = *(const uint4*)pa;          // K = lh*8 .. +7
      fa[mi].q[1] = *(const uint4*)(pa + 16);   // K = 16+lh*8 .. +7
    }
    #pragma unroll
    for (int ni = 0; ni < 2; ++ni) {
      const unsigned short* pb = &lB[(wn + ni * 16 + l16) * LDT + lh * 8];
      fb[ni].q[0] = *(const uint4*)pb;          // K = lh*8 .. +7
      fb[ni].q[1] = *(const uint4*)(pb + 16);   // K = 16+lh*8 .. +7
    }

    #pragma unroll
    for (int mi = 0; mi < 4; ++mi)
      #pragma unroll
      for (int ni = 0; ni < 2; ++ni)
        acc[mi][ni] = __builtin_amdgcn_wmma_f32_16x16x32_bf16(
            false, fa[mi].v, false, fb[ni].v, (short)0, acc[mi][ni], false, false);

    __syncthreads();
  }

  // ------------------------------ epilogue ------------------------------
  const float sc = (MODE == 0) ? p.scale[0] : 1.0f;
  #pragma unroll
  for (int mi = 0; mi < 4; ++mi) {
    #pragma unroll
    for (int ni = 0; ni < 2; ++ni) {
      const int gmb = m0 + wm + mi * 16 + lh * 8;
      const int gn  = n0 + wn + ni * 16 + l16;
      #pragma unroll
      for (int r = 0; r < 8; ++r) {
        const int gm = gmb + r;   // D[m= lh*8 + r][n = lane%16] per ISA C/D layout
        float v = acc[mi][ni][r];
        if (MODE == 0) {
          v = (v + p.b_h2e[gn] + p.x[((size_t)bz * NT + gm) * NE + gn]) * sc;
          p.cc[((size_t)bz * NT + gm) * NE + gn] = f2bf(v);
        } else if (MODE == 1) {
          p.attn[((size_t)bz * NT + gm) * NS + gn] = v;
        } else if (MODE == 2) {
          p.att[((size_t)bz * NT + gm) * NE + gn] = f2bf(v);
        } else {
          const size_t o = ((size_t)bz * NC + gn) * NT + gm;
          p.out2[o] = p.conved[o] + p.b_e2h[gn] + v;
        }
      }
    }
  }
}

// In-place softmax over last dim (S=1024), one 256-thread block per row.
__global__ __launch_bounds__(256) void softmax_rows(float* attn) {
  const int row = blockIdx.x;                     // B*T rows
  float* pr = attn + (size_t)row * NS;
  __shared__ float red[256];
  const int tid = threadIdx.x;

  float m = -3.402823466e38f;
  for (int i = tid; i < NS; i += 256) m = fmaxf(m, pr[i]);
  red[tid] = m; __syncthreads();
  for (int s = 128; s > 0; s >>= 1) {
    if (tid < s) red[tid] = fmaxf(red[tid], red[tid + s]);
    __syncthreads();
  }
  m = red[0]; __syncthreads();

  float sum = 0.f;
  for (int i = tid; i < NS; i += 256) {
    float e = __expf(pr[i] - m);
    pr[i] = e;
    sum += e;
  }
  red[tid] = sum; __syncthreads();
  for (int s = 128; s > 0; s >>= 1) {
    if (tid < s) red[tid] += red[tid + s];
    __syncthreads();
  }
  const float inv = 1.0f / red[0];
  __syncthreads();
  for (int i = tid; i < NS; i += 256) pr[i] *= inv;
}

extern "C" void kernel_launch(void* const* d_in, const int* in_sizes, int n_in,
                              void* d_out, int out_size, void* d_ws, size_t ws_size,
                              hipStream_t stream) {
  Params p;
  p.conved       = (const float*)d_in[0];
  p.enc_conved   = (const float*)d_in[1];
  p.enc_combined = (const float*)d_in[2];
  p.x            = (const float*)d_in[3];
  p.scale        = (const float*)d_in[4];
  p.w_h2e        = (const float*)d_in[5];
  p.b_h2e        = (const float*)d_in[6];
  p.w_e2h        = (const float*)d_in[7];
  p.b_e2h        = (const float*)d_in[8];

  float* out = (float*)d_out;
  p.attn = out;                                   // [B,T,S]
  p.out2 = out + (size_t)NB * NT * NS;            // [B,C,T]
  p.cc   = (unsigned short*)d_ws;                 // 16 MB bf16
  p.att  = p.cc + (size_t)NB * NT * NE;           // 16 MB bf16

  const dim3 blk(256);
  gemm_wmma<0><<<dim3(NT / BM, NE / BN, NB), blk, 0, stream>>>(p);
  gemm_wmma<1><<<dim3(NT / BM, NS / BN, NB), blk, 0, stream>>>(p);
  softmax_rows<<<dim3(NB * NT), blk, 0, stream>>>(p.attn);
  gemm_wmma<2><<<dim3(NT / BM, NE / BN, NB), blk, 0, stream>>>(p);
  gemm_wmma<3><<<dim3(NT / BM, NC / BN, NB), blk, 0, stream>>>(p);
}
